// AttentionStoreActivationPrune_42502996362085
// MI455X (gfx1250) — matrix operand order repaired and checked by reading, not compiled, not run on hardware
//
#include <hip/hip_runtime.h>

typedef __bf16 v16bf __attribute__((ext_vector_type(16)));
typedef float  v8f   __attribute__((ext_vector_type(8)));
typedef unsigned int v4u __attribute__((ext_vector_type(4)));
typedef unsigned int v8u __attribute__((ext_vector_type(8)));

#define S_LEN 577
#define HDIM  768
#define NH    12
#define HD    64
#define BATCH 8
#define MROWS (BATCH * S_LEN)   // 4616
#define SPAD  640               // padded row length of transposed-V global buffer

union FragU { v16bf v; uint4 u[2]; };

// ---------------------------------------------------------------------------
// TDM 2D tile load: builds a D# (group0 128b + group1 256b) in SGPRs and issues
// tensor_load_to_lds.  x = contiguous dim, y = strided dim; OOB reads zero-fill.
// flags = data_size / pad_enable / pad_interval / pad_amount word (group1[31:0]).
// ---------------------------------------------------------------------------
__device__ __forceinline__ void tdm_load_2d(unsigned ldsb, const void* gp, unsigned flags,
                                            unsigned tdim0, unsigned tdim1,
                                            unsigned tile0, unsigned tile1, unsigned stride) {
  unsigned long long ga = (unsigned long long)(size_t)gp;
  v4u g0; v8u g1;
  g0[0] = 1u;                                   // count=1
  g0[1] = ldsb;                                 // lds_addr
  g0[2] = (unsigned)ga;                         // global_addr[31:0]
  g0[3] = (unsigned)(ga >> 32) | 0x80000000u;   // global_addr[56:32] | type=2
  g1[0] = flags;
  g1[1] = (tdim0 & 0xffffu) << 16;                              // tensor_dim0[15:0]
  g1[2] = ((tdim0 >> 16) & 0xffffu) | ((tdim1 & 0xffffu) << 16);// tdim0 hi | tdim1 lo
  g1[3] = ((tdim1 >> 16) & 0xffffu) | ((tile0 & 0xffffu) << 16);// tdim1 hi | tile_dim0
  g1[4] = tile1 & 0xffffu;                                      // tile_dim1 (tile_dim2=0)
  g1[5] = stride;                                               // tensor_dim0_stride[31:0]
  g1[6] = 0u; g1[7] = 0u;
  asm volatile("tensor_load_to_lds %0, %1" :: "s"(g0), "s"(g1) : "memory");
}

// row = 32 bf16 = 16 DWORDs (enc 3), pad 4 DWORDs (enc 3) -> LDS stride 40 elems
#define TDM_FLAGS_K32 ((1u << 16) | (1u << 20) | (3u << 22) | (3u << 25))
// row = 64 bf16 = 32 DWORDs (enc 4), pad 4 DWORDs (enc 3) -> LDS stride 72 elems
#define TDM_FLAGS_K64 ((1u << 16) | (1u << 20) | (4u << 22) | (3u << 25))

// ---------------------------------------------------------------------------
// CDNA5 WMMA bf16 fragment loaders (ISA 7.12.2 layouts), 16B LDS loads.
// ---------------------------------------------------------------------------
__device__ __forceinline__ v16bf frag_a(const __bf16* base, int stride, int lane, int kbase) {
  FragU r;
  const __bf16* p = base + (lane & 15) * stride + kbase + ((lane >> 4) << 3);
  r.u[0] = *(const uint4*)(p);
  r.u[1] = *(const uint4*)(p + 16);
  return r.v;
}
__device__ __forceinline__ v16bf frag_b(const __bf16* base, int stride, int lane, int kbase) {
  FragU r;
  const __bf16* p = base + (lane & 15) * stride + kbase + ((lane >> 4) << 3);
  r.u[0] = *(const uint4*)(p);
  r.u[1] = *(const uint4*)(p + 16);
  return r.v;
}

// ---------------------------------------------------------------------------
// fp32 -> bf16, 4 elements/thread
// ---------------------------------------------------------------------------
__global__ void cvt_bf16(const float* __restrict__ x, __bf16* __restrict__ y, int n4) {
  int i = blockIdx.x * 256 + threadIdx.x;
  if (i < n4) {
    float4 f = ((const float4*)x)[i];
    union { __bf16 h[4]; unsigned long long u; } o;
    o.h[0] = (__bf16)f.x; o.h[1] = (__bf16)f.y;
    o.h[2] = (__bf16)f.z; o.h[3] = (__bf16)f.w;
    ((unsigned long long*)y)[i] = o.u;
  }
}

// fp32 W[k][n] -> bf16 Wt[n][k]  (768x768)
__global__ void cvt_transpose_w(const float* __restrict__ W, __bf16* __restrict__ Wt) {
  int i = blockIdx.x * 256 + threadIdx.x;
  if (i < HDIM * HDIM) {
    int k = i / HDIM, n = i % HDIM;
    Wt[(size_t)n * HDIM + k] = (__bf16)W[i];
  }
}

// ---------------------------------------------------------------------------
// C[M,768] = A[M,768]*W + bias ; W given N-major bf16 (Wt[n][k]).
// Block tile 128x128, 8 waves as 4(M)x2(N).  Both LDS tiles are fetched by the
// Tensor Data Mover (wave 0 issues 2 descriptors per K-step, double-buffered).
// MODE: 0 = bf16 row-major out, 1 = f32 row-major out, 2 = bf16 transposed-V out.
// ---------------------------------------------------------------------------
template <int MODE>
__global__ __launch_bounds__(256) void gemm_x_wt(const __bf16* __restrict__ A,
                                                 const __bf16* __restrict__ Bt,
                                                 const float* __restrict__ bias,
                                                 void* __restrict__ outp, int M) {
  __shared__ __align__(16) __bf16 As[2][128 * 40];  // stride 40 elems (TDM-padded)
  __shared__ __align__(16) __bf16 Bs[2][128 * 40];

  const int tid = threadIdx.x, lane = tid & 31, wave = tid >> 5;
  const int wm = wave >> 1, wn = wave & 1;
  const int mBase = blockIdx.y * 128, nBase = blockIdx.x * 128;

  v8f acc[2][4];
  for (int i = 0; i < 2; i++)
    for (int j = 0; j < 4; j++)
      for (int r = 0; r < 8; r++) acc[i][j][r] = 0.f;

  const unsigned mRem = (unsigned)(M - mBase);   // tail rows zero-filled by TDM

  auto issue = [&](int k0, int buf) {
    if (wave == 0) {
      tdm_load_2d((unsigned)(size_t)(void*)&As[buf][0], A + (size_t)mBase * HDIM + k0,
                  TDM_FLAGS_K32, 32u, mRem, 32u, 128u, HDIM);
      tdm_load_2d((unsigned)(size_t)(void*)&Bs[buf][0], Bt + (size_t)nBase * HDIM + k0,
                  TDM_FLAGS_K32, 32u, 128u, 32u, 128u, HDIM);
    }
  };

  issue(0, 0);
  if (wave == 0) __builtin_amdgcn_s_wait_tensorcnt(0);
  __syncthreads();

  int buf = 0;
  for (int k0 = 0; k0 < HDIM; k0 += 32, buf ^= 1) {
    if (k0 + 32 < HDIM) issue(k0 + 32, buf ^ 1);   // DMA next tiles under WMMAs

    v16bf af[2], bfr[4];
    for (int i = 0; i < 2; i++) af[i] = frag_a(&As[buf][(wm * 32 + i * 16) * 40], 40, lane, 0);
    for (int j = 0; j < 4; j++) bfr[j] = frag_b(&Bs[buf][(wn * 64 + j * 16) * 40], 40, lane, 0);
    for (int i = 0; i < 2; i++)
      for (int j = 0; j < 4; j++)
        acc[i][j] = __builtin_amdgcn_wmma_f32_16x16x32_bf16(
            false, af[i], false, bfr[j], (short)0, acc[i][j], false, false);

    if (wave == 0 && k0 + 32 < HDIM) __builtin_amdgcn_s_wait_tensorcnt(0);
    __syncthreads();
  }

  for (int i = 0; i < 2; i++) {
    int mrow = mBase + wm * 32 + i * 16 + ((lane >> 4) << 3);
    for (int j = 0; j < 4; j++) {
      int n = nBase + wn * 64 + j * 16 + (lane & 15);
      float bv = bias[n];
      for (int r = 0; r < 8; r++) {
        int mm = mrow + r;
        if (mm < M) {
          float val = acc[i][j][r] + bv;
          if (MODE == 1) {
            ((float*)outp)[(size_t)mm * HDIM + n] = val;
          } else if (MODE == 0) {
            ((__bf16*)outp)[(size_t)mm * HDIM + n] = (__bf16)val;
          } else {  // transposed V: Vt[b*768 + n][s], row stride SPAD
            int bb = mm / S_LEN, ss = mm - bb * S_LEN;
            ((__bf16*)outp)[((size_t)(bb * HDIM + n)) * SPAD + ss] = (__bf16)val;
          }
        }
      }
    }
  }
}

// ---------------------------------------------------------------------------
// Flash attention: block = (qtile of 64 rows, head, batch), 4 waves x 16 q-rows.
// Both K and V tiles are TDM-loaded (V from the transposed global buffer),
// double-buffered so chunk ct+1 streams in under chunk ct's WMMAs/softmax.
// TDM zero-fills the s >= 577 tails of both tiles.
// ---------------------------------------------------------------------------
__global__ __launch_bounds__(128) void flash_attn(const __bf16* __restrict__ Q,
                                                  const __bf16* __restrict__ K,
                                                  const __bf16* __restrict__ Vt_g,
                                                  __bf16* __restrict__ O) {
  __shared__ __align__(16) __bf16 Kt[2][64 * 72];   // [kcol][d], row 128B + 16B pad
  __shared__ __align__(16) __bf16 Vt[2][64 * 72];   // [d][kcol]
  __shared__ __align__(16) __bf16 Ps[4][16 * 72];   // per-wave P round-trip buffer

  const int tid = threadIdx.x, lane = tid & 31, wave = tid >> 5;
  const int qt = blockIdx.x, h = blockIdx.y, b = blockIdx.z;
  const int q0 = qt * 64 + wave * 16;

  // Persistent Q A-fragments (d-chunks 0..31 and 32..63), rows clamped at S.
  int sr = q0 + (lane & 15); if (sr >= S_LEN) sr = S_LEN - 1;
  const __bf16* qp = Q + ((size_t)(b * S_LEN + sr)) * HDIM + h * HD;
  v16bf aq[2];
  {
    FragU f0, f1;
    int off = (lane >> 4) << 3;
    f0.u[0] = *(const uint4*)(qp + off);
    f0.u[1] = *(const uint4*)(qp + off + 16);
    f1.u[0] = *(const uint4*)(qp + 32 + off);
    f1.u[1] = *(const uint4*)(qp + 32 + off + 16);
    aq[0] = f0.v; aq[1] = f1.v;
  }

  v8f o_acc[4];
  for (int t = 0; t < 4; t++)
    for (int r = 0; r < 8; r++) o_acc[t][r] = 0.f;
  float m8[8], l8[8];
  for (int i = 0; i < 8; i++) { m8[i] = -3.0e38f; l8[i] = 0.f; }

  auto issueKV = [&](int ct, int buf) {
    if (wave == 0) {
      unsigned nrows = (unsigned)(S_LEN - ct * 64); if (nrows > 64u) nrows = 64u;
      // K chunk: x = d (64 valid), y = s rows (tail zero-filled)
      tdm_load_2d((unsigned)(size_t)(void*)&Kt[buf][0],
                  K + ((size_t)(b * S_LEN + ct * 64)) * HDIM + h * HD,
                  TDM_FLAGS_K64, 64u, nrows, 64u, 64u, HDIM);
      // V chunk from transposed buffer: x = s (tail zero-filled), y = d
      tdm_load_2d((unsigned)(size_t)(void*)&Vt[buf][0],
                  Vt_g + ((size_t)((b * NH + h) * HD)) * SPAD + ct * 64,
                  TDM_FLAGS_K64, nrows, 64u, 64u, 64u, SPAD);
    }
  };

  issueKV(0, 0);
  if (wave == 0) __builtin_amdgcn_s_wait_tensorcnt(0);
  __syncthreads();

  int buf = 0;
  for (int ct = 0; ct < 10; ++ct, buf ^= 1) {
    if (ct + 1 < 10) issueKV(ct + 1, buf ^ 1);   // stream next chunk under compute

    // ---- scores: S = Q K^T / 8 (batch 4 B-frags, then 4 WMMAs) ----
    v8f s[4];
    {
      v16bf bk[4];
      for (int j = 0; j < 4; j++) bk[j] = frag_b(&Kt[buf][(j * 16) * 72], 72, lane, 0);
      for (int j = 0; j < 4; j++) {
        v8f z;
        for (int r = 0; r < 8; r++) z[r] = 0.f;
        s[j] = __builtin_amdgcn_wmma_f32_16x16x32_bf16(false, aq[0], false, bk[j], (short)0, z, false, false);
      }
      for (int j = 0; j < 4; j++) bk[j] = frag_b(&Kt[buf][(j * 16) * 72], 72, lane, 32);
      for (int j = 0; j < 4; j++)
        s[j] = __builtin_amdgcn_wmma_f32_16x16x32_bf16(false, aq[1], false, bk[j], (short)0, s[j], false, false);
    }
    for (int j = 0; j < 4; j++) {
      int col = ct * 64 + j * 16 + (lane & 15);
      bool oob = (col >= S_LEN);
      for (int r = 0; r < 8; r++) s[j][r] = oob ? -3.0e38f : s[j][r] * 0.125f;
    }

    // ---- online softmax (row reductions across 16-lane halves) ----
    for (int i = 0; i < 8; i++) {
      float vmax = fmaxf(fmaxf(s[0][i], s[1][i]), fmaxf(s[2][i], s[3][i]));
      for (int d = 1; d < 16; d <<= 1) vmax = fmaxf(vmax, __shfl_xor(vmax, d, 32));
      float nm = fmaxf(m8[i], vmax);
      float alpha = __expf(m8[i] - nm);
      float rsum = 0.f;
      for (int j = 0; j < 4; j++) {
        float p = __expf(s[j][i] - nm);
        s[j][i] = p;
        rsum += p;
      }
      for (int d = 1; d < 16; d <<= 1) rsum += __shfl_xor(rsum, d, 32);
      l8[i] = l8[i] * alpha + rsum;
      m8[i] = nm;
      for (int t = 0; t < 4; t++) o_acc[t][i] *= alpha;
    }

    // ---- C-layout -> A-layout via per-wave LDS buffer ----
    __bf16* pw = &Ps[wave][0];
    {
      int mrow = (lane >> 4) << 3;
      for (int j = 0; j < 4; j++)
        for (int r = 0; r < 8; r++)
          pw[(mrow + r) * 72 + j * 16 + (lane & 15)] = (__bf16)s[j][r];
    }
    asm volatile("s_wait_dscnt 0" ::: "memory");  // same-wave LDS RAW

    // ---- O += P * V (batch 4 B-frags, then 4 WMMAs, twice) ----
    v16bf pa0 = frag_a(pw, 72, lane, 0);
    v16bf pa1 = frag_a(pw, 72, lane, 32);
    {
      v16bf bv[4];
      for (int t = 0; t < 4; t++) bv[t] = frag_b(&Vt[buf][(t * 16) * 72], 72, lane, 0);
      for (int t = 0; t < 4; t++)
        o_acc[t] = __builtin_amdgcn_wmma_f32_16x16x32_bf16(false, pa0, false, bv[t], (short)0, o_acc[t], false, false);
      for (int t = 0; t < 4; t++) bv[t] = frag_b(&Vt[buf][(t * 16) * 72], 72, lane, 32);
      for (int t = 0; t < 4; t++)
        o_acc[t] = __builtin_amdgcn_wmma_f32_16x16x32_bf16(false, pa1, false, bv[t], (short)0, o_acc[t], false, false);
    }

    if (wave == 0 && ct + 1 < 10) __builtin_amdgcn_s_wait_tensorcnt(0);
    __syncthreads();
  }

  for (int t = 0; t < 4; t++) {
    int dcol = h * HD + t * 16 + (lane & 15);
    for (int r = 0; r < 8; r++) {
      int srow = q0 + ((lane >> 4) << 3) + r;
      if (srow < S_LEN) {
        float val = o_acc[t][r] / l8[r];
        O[((size_t)(b * S_LEN + srow)) * HDIM + dcol] = (__bf16)val;
      }
    }
  }
}

// ---------------------------------------------------------------------------
extern "C" void kernel_launch(void* const* d_in, const int* in_sizes, int n_in,
                              void* d_out, int out_size, void* d_ws, size_t ws_size,
                              hipStream_t stream) {
  const float* hs = (const float*)d_in[0];
  const float* Wq = (const float*)d_in[1]; const float* bq = (const float*)d_in[2];
  const float* Wk = (const float*)d_in[3]; const float* bk = (const float*)d_in[4];
  const float* Wv = (const float*)d_in[5]; const float* bv = (const float*)d_in[6];
  const float* Wo = (const float*)d_in[7]; const float* bo = (const float*)d_in[8];

  const size_t MH = (size_t)MROWS * HDIM;        // 3,545,088 elems
  const size_t WW = (size_t)HDIM * HDIM;         //   589,824 elems
  const size_t VT = (size_t)BATCH * HDIM * SPAD; // transposed-V buffer elems
  size_t off = 0;
  auto take = [&](size_t bytes) -> void* {
    void* p = (char*)d_ws + off;
    off += (bytes + 255) & ~(size_t)255;
    return p;
  };
  __bf16* xb  = (__bf16*)take(MH * 2);
  __bf16* Wqt = (__bf16*)take(WW * 2);
  __bf16* Wkt = (__bf16*)take(WW * 2);
  __bf16* Wvt = (__bf16*)take(WW * 2);
  __bf16* Wot = (__bf16*)take(WW * 2);
  __bf16* Qb  = (__bf16*)take(MH * 2);
  __bf16* Kb  = (__bf16*)take(MH * 2);
  __bf16* Vtb = (__bf16*)take(VT * 2);
  __bf16* Cb  = (__bf16*)take(MH * 2);

  // 1. precision conversion / weight transpose
  int n4 = (int)(MH / 4);
  cvt_bf16<<<(n4 + 255) / 256, 256, 0, stream>>>(hs, xb, n4);
  int wgrid = (int)((WW + 255) / 256);
  cvt_transpose_w<<<wgrid, 256, 0, stream>>>(Wq, Wqt);
  cvt_transpose_w<<<wgrid, 256, 0, stream>>>(Wk, Wkt);
  cvt_transpose_w<<<wgrid, 256, 0, stream>>>(Wv, Wvt);
  cvt_transpose_w<<<wgrid, 256, 0, stream>>>(Wo, Wot);

  // 2. QKV projections (Q,K bf16 row-major; V bf16 transposed layout)
  dim3 ggrid(HDIM / 128, (MROWS + 127) / 128);   // 6 x 37
  gemm_x_wt<0><<<ggrid, 256, 0, stream>>>(xb, Wqt, bq, Qb, MROWS);
  gemm_x_wt<0><<<ggrid, 256, 0, stream>>>(xb, Wkt, bk, Kb, MROWS);
  gemm_x_wt<2><<<ggrid, 256, 0, stream>>>(xb, Wvt, bv, Vtb, MROWS);

  // 3. attention
  dim3 fgrid((S_LEN + 63) / 64, NH, BATCH);      // 10 x 12 x 8
  flash_attn<<<fgrid, 128, 0, stream>>>(Qb, Kb, Vtb, Cb);

  // 4. output projection (fp32 out, + bias)
  gemm_x_wt<1><<<ggrid, 256, 0, stream>>>(Cb, Wot, bo, (float*)d_out, MROWS);
}
